// S4Model_30365418783277
// MI455X (gfx1250) — compile-verified, matrix operand order checked by reading, not run on hardware
//
#include <hip/hip_runtime.h>

// ---------------------------------------------------------------------------
// S4D model for MI455X (gfx1250, wave32).
//  - GEMMs: v_wmma_f32_16x16x32_bf16, N-blocked per wave so one A fragment
//    feeds 4 WMMAs (L2-traffic reduction; GEMMs stay HBM-bound, not compute)
//  - sequence mixing: chunked two-pass parallel scan (32 chunks of 128),
//    65536-thread passes instead of a 2048-thread serial recurrence
//  - GELU fused into scan emit; GLU+bias+residual fused into GEMM epilogue
// ---------------------------------------------------------------------------

typedef __attribute__((ext_vector_type(16))) __bf16 v16bf;
typedef __attribute__((ext_vector_type(8)))  float  v8f;

#define B_   8
#define L_   4096
#define DIN_ 128
#define H_   256
#define N2_  8
#define NL_  4
#define MTOT (B_ * L_)          // 32768 rows in all GEMMs
#define NC_  32                 // scan chunks
#define CL_  128                // chunk length (NC_*CL_ == L_)

// Load one lane's slice of a 16x32 bf16 WMMA fragment from a row-major fp32
// matrix (leading dim `ld`). Per ISA 7.12.2: lanes 0-15 hold K = k0+0..7 and
// k0+16..23; lanes 16-31 hold K = k0+8..15 and k0+24..31. Two pairs of
// contiguous float4 loads (global_load_b128), converted to bf16 in registers.
__device__ __forceinline__ v16bf load_frag_bf16(const float* __restrict__ base,
                                                int ld, int row, int k0, int lane) {
    const int kbase = (lane & 16) >> 1;                       // 0 or 8
    const float* p = base + (size_t)row * ld + (k0 + kbase);  // 16B aligned
    float4 f0 = *reinterpret_cast<const float4*>(p);
    float4 f1 = *reinterpret_cast<const float4*>(p + 4);
    float4 f2 = *reinterpret_cast<const float4*>(p + 16);
    float4 f3 = *reinterpret_cast<const float4*>(p + 20);
    v16bf r;
    r[0]  = (__bf16)f0.x;  r[1]  = (__bf16)f0.y;  r[2]  = (__bf16)f0.z;  r[3]  = (__bf16)f0.w;
    r[4]  = (__bf16)f1.x;  r[5]  = (__bf16)f1.y;  r[6]  = (__bf16)f1.z;  r[7]  = (__bf16)f1.w;
    r[8]  = (__bf16)f2.x;  r[9]  = (__bf16)f2.y;  r[10] = (__bf16)f2.z;  r[11] = (__bf16)f2.w;
    r[12] = (__bf16)f3.x;  r[13] = (__bf16)f3.y;  r[14] = (__bf16)f3.z;  r[15] = (__bf16)f3.w;
    return r;
}

// Per-(h,n) discrete pole w = exp(dt*A), A = -exp(log_A_real) + i*A_imag.
__device__ __forceinline__ void s4d_pole(const float* __restrict__ log_A_real,
                                         const float* __restrict__ A_imag,
                                         int hh, int n, float dt,
                                         float& Ar, float& Ai, float& wr, float& wi) {
    Ar = -__expf(log_A_real[hh * N2_ + n]);
    Ai = A_imag[hh * N2_ + n];
    const float er = __expf(Ar * dt);
    wr = er * __cosf(Ai * dt);
    wi = er * __sinf(Ai * dt);
}

// ---------------------------------------------------------------------------
// Encoder: h = x @ enc_w^T + enc_b.   (32768 x 128)·(128 x 256)
// Each wave owns a 16(M) x 64(N) tile: one A fragment -> 4 WMMAs per k-step.
// ---------------------------------------------------------------------------
__global__ void __launch_bounds__(256)
enc_gemm(const float* __restrict__ x, const float* __restrict__ w,
         const float* __restrict__ bias, float* __restrict__ h) {
    const int wave = (blockIdx.x * blockDim.x + threadIdx.x) >> 5;
    const int mt   = wave >> 2;          // 2048 M tiles
    const int ng   = wave & 3;           // 4 N groups of 64 cols
    const int lane = threadIdx.x & 31;
    const int row0 = mt * 16, col0 = ng * 64;
    const int am = row0 + (lane & 15);

    v8f acc[4] = {{}, {}, {}, {}};
    #pragma unroll
    for (int k0 = 0; k0 < DIN_; k0 += 32) {
        v16bf a = load_frag_bf16(x, DIN_, am, k0, lane);
        #pragma unroll
        for (int j = 0; j < 4; ++j) {
            const int bn = col0 + j * 16 + (lane & 15);
            v16bf b = load_frag_bf16(w, DIN_, bn, k0, lane);
            acc[j] = __builtin_amdgcn_wmma_f32_16x16x32_bf16(
                false, a, false, b, (short)0, acc[j], false, false);
        }
    }
    const int rbase = row0 + ((lane >> 4) << 3);
    #pragma unroll
    for (int j = 0; j < 4; ++j) {
        const int   col  = col0 + j * 16 + (lane & 15);
        const float bcol = bias[col];
        #pragma unroll
        for (int r = 0; r < 8; ++r)
            h[(size_t)(rbase + r) * H_ + col] = acc[j][r] + bcol;
    }
}

// ---------------------------------------------------------------------------
// Scan pass 1: per-(b,chunk,h) local chunk scan from zero state; store the
// chunk-end state (8 complex values).  65536 threads, coalesced along h.
// ---------------------------------------------------------------------------
__global__ void __launch_bounds__(256)
scan_chunk_local(const float* __restrict__ hbuf,
                 const float* __restrict__ log_dt,
                 const float* __restrict__ log_A_real,
                 const float* __restrict__ A_imag,
                 float* __restrict__ states) {
    const int tid = blockIdx.x * blockDim.x + threadIdx.x;   // 0..65535
    const int hh  = tid & (H_ - 1);
    const int c   = (tid >> 8) & (NC_ - 1);
    const int b   = tid >> 13;

    const float dt = __expf(log_dt[hh]);
    float wre[N2_], wim[N2_];
    #pragma unroll
    for (int n = 0; n < N2_; ++n) {
        float Ar, Ai;
        s4d_pole(log_A_real, A_imag, hh, n, dt, Ar, Ai, wre[n], wim[n]);
    }
    float sre[N2_], sim[N2_];
    #pragma unroll
    for (int n = 0; n < N2_; ++n) { sre[n] = 0.0f; sim[n] = 0.0f; }

    const float* up = hbuf + ((size_t)b * L_ + (size_t)c * CL_) * H_ + hh;
    for (int l = 0; l < CL_; ++l) {
        const float u = up[(size_t)l * H_];
        #pragma unroll
        for (int n = 0; n < N2_; ++n) {
            const float sr = wre[n] * sre[n] - wim[n] * sim[n] + u;
            const float si = wre[n] * sim[n] + wim[n] * sre[n];
            sre[n] = sr;  sim[n] = si;
        }
    }
    float* st = states + ((size_t)b * NC_ + c) * (N2_ * 2) * H_ + hh;
    #pragma unroll
    for (int n = 0; n < N2_; ++n) {
        st[(size_t)(2 * n + 0) * H_] = sre[n];
        st[(size_t)(2 * n + 1) * H_] = sim[n];
    }
}

// ---------------------------------------------------------------------------
// Scan pass 2: per-(b,h) carry recurrence across the 32 chunks using the
// exact chunk decay w^CL = exp(CL*dt*A).  Overwrites states[] in place with
// each chunk's carry-IN state.  Only 32 serial steps.
// ---------------------------------------------------------------------------
__global__ void __launch_bounds__(256)
scan_carry(const float* __restrict__ log_dt,
           const float* __restrict__ log_A_real,
           const float* __restrict__ A_imag,
           float* __restrict__ states) {
    const int tid = blockIdx.x * blockDim.x + threadIdx.x;   // 0..2047
    const int hh  = tid & (H_ - 1);
    const int b   = tid >> 8;

    const float dt = __expf(log_dt[hh]);
    float wTr[N2_], wTi[N2_];
    #pragma unroll
    for (int n = 0; n < N2_; ++n) {
        const float Ar = -__expf(log_A_real[hh * N2_ + n]);
        const float Ai = A_imag[hh * N2_ + n];
        const float er = __expf(Ar * dt * (float)CL_);
        wTr[n] = er * __cosf(Ai * dt * (float)CL_);
        wTi[n] = er * __sinf(Ai * dt * (float)CL_);
    }
    float er_[N2_], ei_[N2_];   // running s_end
    #pragma unroll
    for (int n = 0; n < N2_; ++n) { er_[n] = 0.0f; ei_[n] = 0.0f; }

    for (int c = 0; c < NC_; ++c) {
        float* st = states + ((size_t)b * NC_ + c) * (N2_ * 2) * H_ + hh;
        #pragma unroll
        for (int n = 0; n < N2_; ++n) {
            const float Sr = st[(size_t)(2 * n + 0) * H_];   // local chunk sum
            const float Si = st[(size_t)(2 * n + 1) * H_];
            st[(size_t)(2 * n + 0) * H_] = er_[n];           // carry-in(c)
            st[(size_t)(2 * n + 1) * H_] = ei_[n];
            const float nr = wTr[n] * er_[n] - wTi[n] * ei_[n] + Sr;
            const float ni = wTr[n] * ei_[n] + wTi[n] * er_[n] + Si;
            er_[n] = nr;  ei_[n] = ni;
        }
    }
}

// ---------------------------------------------------------------------------
// Scan pass 3: replay each chunk from its carry-in; emit
// gelu( 2*Re(sum_n Ck_n * s_n) + D_skip*u ).  65536 threads.
// ---------------------------------------------------------------------------
__global__ void __launch_bounds__(256)
scan_emit(const float* __restrict__ hbuf,
          const float* __restrict__ log_dt,
          const float* __restrict__ log_A_real,
          const float* __restrict__ A_imag,
          const float* __restrict__ C_re,
          const float* __restrict__ C_im,
          const float* __restrict__ D_skip,
          const float* __restrict__ states,
          float* __restrict__ yact) {
    const int tid = blockIdx.x * blockDim.x + threadIdx.x;   // 0..65535
    const int hh  = tid & (H_ - 1);
    const int c   = (tid >> 8) & (NC_ - 1);
    const int b   = tid >> 13;

    const float dt = __expf(log_dt[hh]);
    float wre[N2_], wim[N2_], cr2[N2_], ci2[N2_];
    #pragma unroll
    for (int n = 0; n < N2_; ++n) {
        float Ar, Ai;
        s4d_pole(log_A_real, A_imag, hh, n, dt, Ar, Ai, wre[n], wim[n]);
        // Ck = (C_re + i C_im) * (w - 1) / A ;  fold the output factor of 2.
        const float den = Ar * Ar + Ai * Ai;
        const float nr = wre[n] - 1.0f, ni = wim[n];
        const float qr = (nr * Ar + ni * Ai) / den;
        const float qi = (ni * Ar - nr * Ai) / den;
        const float cr = C_re[hh * N2_ + n], ci = C_im[hh * N2_ + n];
        cr2[n] = 2.0f * (cr * qr - ci * qi);
        ci2[n] = 2.0f * (cr * qi + ci * qr);
    }
    const float dsk = D_skip[hh];

    float sre[N2_], sim[N2_];
    const float* st = states + ((size_t)b * NC_ + c) * (N2_ * 2) * H_ + hh;
    #pragma unroll
    for (int n = 0; n < N2_; ++n) {
        sre[n] = st[(size_t)(2 * n + 0) * H_];   // carry-in from pass 2
        sim[n] = st[(size_t)(2 * n + 1) * H_];
    }

    const size_t off = ((size_t)b * L_ + (size_t)c * CL_) * H_ + hh;
    const float* up = hbuf + off;
    float*       yp = yact + off;
    for (int l = 0; l < CL_; ++l) {
        const float u = up[(size_t)l * H_];
        float y = dsk * u;
        #pragma unroll
        for (int n = 0; n < N2_; ++n) {
            const float sr = wre[n] * sre[n] - wim[n] * sim[n] + u;
            const float si = wre[n] * sim[n] + wim[n] * sre[n];
            sre[n] = sr;  sim[n] = si;
            y += cr2[n] * sr - ci2[n] * si;      // 2*Re(Ck * s)
        }
        // GELU (tanh approximation, matches jax.nn.gelu default)
        const float t = tanhf(0.7978845608f * (y + 0.044715f * y * y * y));
        yp[(size_t)l * H_] = 0.5f * y * (1.0f + t);
    }
}

// ---------------------------------------------------------------------------
// 1x1 conv H->2H + GLU + residual: (32768 x 256)·(256 x 512), fused epilogue.
// Each wave owns 2 N-tiles and both their "a" (cols p) and "g" (cols p+256)
// accumulators: one A fragment feeds 4 WMMAs; 2H intermediate never hits
// memory.
// ---------------------------------------------------------------------------
__global__ void __launch_bounds__(256)
outproj_glu(const float* __restrict__ yact, const float* __restrict__ ow,
            const float* __restrict__ ob, float* __restrict__ hbuf) {
    const int wave = (blockIdx.x * blockDim.x + threadIdx.x) >> 5;
    const int mt   = wave >> 3;          // 2048 M tiles
    const int ng   = wave & 7;           // 8 N groups of 32 "a" cols
    const int lane = threadIdx.x & 31;
    const int row0 = mt * 16, col0 = ng * 32;
    const int am = row0 + (lane & 15);

    v8f acc_a[2] = {{}, {}}, acc_g[2] = {{}, {}};
    #pragma unroll
    for (int k0 = 0; k0 < H_; k0 += 32) {
        v16bf a = load_frag_bf16(yact, H_, am, k0, lane);
        #pragma unroll
        for (int j = 0; j < 2; ++j) {
            const int pa = col0 + j * 16 + (lane & 15);
            v16bf ba = load_frag_bf16(ow, H_, pa,      k0, lane);
            v16bf bg = load_frag_bf16(ow, H_, pa + H_, k0, lane);
            acc_a[j] = __builtin_amdgcn_wmma_f32_16x16x32_bf16(
                false, a, false, ba, (short)0, acc_a[j], false, false);
            acc_g[j] = __builtin_amdgcn_wmma_f32_16x16x32_bf16(
                false, a, false, bg, (short)0, acc_g[j], false, false);
        }
    }
    const int rbase = row0 + ((lane >> 4) << 3);
    #pragma unroll
    for (int j = 0; j < 2; ++j) {
        const int   p     = col0 + j * 16 + (lane & 15);
        const float biasa = ob[p], biasg = ob[p + H_];
        #pragma unroll
        for (int r = 0; r < 8; ++r) {
            const float av  = acc_a[j][r] + biasa;
            const float gv  = acc_g[j][r] + biasg;
            const float glu = av / (1.0f + __expf(-gv));
            const size_t idx = (size_t)(rbase + r) * H_ + p;
            hbuf[idx] += glu;             // residual (h lives in d_out)
        }
    }
}

// ---------------------------------------------------------------------------
extern "C" void kernel_launch(void* const* d_in, const int* in_sizes, int n_in,
                              void* d_out, int out_size, void* d_ws, size_t ws_size,
                              hipStream_t stream) {
    const float* x          = (const float*)d_in[0];
    const float* enc_w      = (const float*)d_in[1];
    const float* enc_b      = (const float*)d_in[2];
    const float* log_dt     = (const float*)d_in[3];
    const float* log_A_real = (const float*)d_in[4];
    const float* A_imag     = (const float*)d_in[5];
    const float* C_re       = (const float*)d_in[6];
    const float* C_im       = (const float*)d_in[7];
    const float* D_skip     = (const float*)d_in[8];
    const float* out_w      = (const float*)d_in[9];
    const float* out_b      = (const float*)d_in[10];

    float* h      = (float*)d_out;                       // (B, L, H) running activation
    float* yact   = (float*)d_ws;                        // (B, L, H) 32 MB
    float* states = yact + (size_t)B_ * L_ * H_;         // (B, NC, N2, 2, H) 4 MB

    // encoder: 2048 M-tiles * 4 N-groups = 8192 waves, 8 waves/block
    enc_gemm<<<1024, 256, 0, stream>>>(x, enc_w, enc_b, h);

    for (int i = 0; i < NL_; ++i) {
        const float* ldt = log_dt     + (size_t)i * H_;
        const float* lar = log_A_real + (size_t)i * H_ * N2_;
        const float* aim = A_imag     + (size_t)i * H_ * N2_;
        scan_chunk_local<<<(B_ * NC_ * H_) / 256, 256, 0, stream>>>(
            h, ldt, lar, aim, states);
        scan_carry<<<(B_ * H_) / 256, 256, 0, stream>>>(
            ldt, lar, aim, states);
        scan_emit<<<(B_ * NC_ * H_) / 256, 256, 0, stream>>>(
            h, ldt, lar, aim,
            C_re + (size_t)i * H_ * N2_,
            C_im + (size_t)i * H_ * N2_,
            D_skip + (size_t)i * H_,
            states, yact);
        // out-proj: 2048 M-tiles * 8 N-groups = 16384 waves, 8 waves/block
        outproj_glu<<<2048, 256, 0, stream>>>(
            yact,
            out_w + (size_t)i * 2 * H_ * H_,
            out_b + (size_t)i * 2 * H_,
            h);
    }
}
